// StarEQueryEmbeddingModel_43533788512909
// MI455X (gfx1250) — matrix operand order repaired and checked by reading, not compiled, not run on hardware
//
#include <hip/hip_runtime.h>
#include <hip/hip_bf16.h>

// ---------------------------------------------------------------------------
// StarE query embedding, CDNA5 (gfx1250) implementation.
// Heavy GEMMs (row-blocks x 256x256 weights) run on v_wmma_f32_16x16x32_bf16
// with f32 accumulation; gathers/scatters use f32 + global_atomic_add_f32.
// qsum tile staging uses GLOBAL_LOAD_ASYNC_TO_LDS_B128 (ASYNCcnt path).
// ---------------------------------------------------------------------------

#define Nn 131072   // nodes
#define Ee 262144   // edges
#define Qq 262144   // qualifiers
#define Gg 16384    // graphs
#define Dd 256      // hidden dim
#define NRr 500     // relations (x_r_table has 2*NRr rows)
#define LL 2        // layers
#define ALPHA_F 0.8f

typedef __attribute__((ext_vector_type(16))) __bf16 v16bf;
typedef __attribute__((ext_vector_type(8)))  float  v8f;

// ---------------------------------------------------------------------------
// Paired WMMA GEMM: two 16x16 f32 C tiles = A(16x256 f32 LDS) @ Wp[nt0|nt1].
// A fragment is built ONCE per K-step and fed to both WMMAs (B differs).
// Wp is pre-swizzled bf16 fragments: element [((nt*8+ks)*32 + lane)*16 + j]
//   = W[k][n], n = nt*16 + (lane&15), k = ks*32 + ((lane>>4)<<4) + j
// A fragment (16-bit A 16x32): lanes 0-15 -> K {kb..kb+15} with kb=0;
// lanes 16-31 same with kb=16 (matches Wp contiguous k order).
// ---------------------------------------------------------------------------
__device__ __forceinline__ void tile_gemm_pair(const float* __restrict__ Alds,
                                               const v16bf* __restrict__ Wp,
                                               int nt0, int nt1, int lane,
                                               v8f& acc0, v8f& acc1) {
#pragma unroll
    for (int r = 0; r < 8; ++r) { acc0[r] = 0.0f; acc1[r] = 0.0f; }
#pragma unroll
    for (int ks = 0; ks < 8; ++ks) {
        const int m  = lane & 15;
        const int kb = (lane >> 4) << 4;
        const float* rp = Alds + m * Dd + ks * 32 + kb;
        v16bf a;
#pragma unroll
        for (int j = 0; j < 16; ++j) {
            a[j]     = (__bf16)rp[j];
        }
        v16bf b0 = Wp[(nt0 * 8 + ks) * 32 + lane];
        v16bf b1 = Wp[(nt1 * 8 + ks) * 32 + lane];
        acc0 = __builtin_amdgcn_wmma_f32_16x16x32_bf16(
            false, a, false, b0, (short)0, acc0, false, false);
        acc1 = __builtin_amdgcn_wmma_f32_16x16x32_bf16(
            false, a, false, b1, (short)0, acc1, false, false);
    }
}

// ---------------------------------------------------------------------------
// Small utility kernels
// ---------------------------------------------------------------------------
__global__ void zero_kernel(float4* __restrict__ p, long long n4) {
    long long t = (long long)blockIdx.x * 256 + threadIdx.x;
    if (t < n4) p[t] = make_float4(0.f, 0.f, 0.f, 0.f);
}

// Convert one 256x256 f32 weight (row-major, W[k*256+n]) to bf16 fragment layout.
__global__ void prep_w_kernel(const float* __restrict__ W, __bf16* __restrict__ dst) {
    int t    = blockIdx.x * 256 + threadIdx.x;       // 65536 total
    int j    = t & 15;
    int lane = (t >> 4) & 31;
    int ks   = (t >> 9) & 7;
    int nt   = t >> 12;
    int n    = nt * 16 + (lane & 15);
    int k    = ks * 32 + ((lane >> 4) << 4) + j;
    dst[t]   = (__bf16)W[k * Dd + n];
}

// x_e = x_e_table[entity_ids]
__global__ void gather_xe_kernel(const float* __restrict__ table,
                                 const int* __restrict__ eid,
                                 float* __restrict__ xe) {
    long long t = (long long)blockIdx.x * 256 + threadIdx.x;  // Nn*64
    int n = (int)(t >> 6);
    int c = ((int)t & 63) << 2;
    float4 v = *(const float4*)(table + (size_t)eid[n] * Dd + c);
    *(float4*)(xe + (size_t)n * Dd + c) = v;
}

// Precompute small relation ids: esid[e] = relation_ids[edge_type[e]],
// qsid[q] = relation_ids[q_rel % E] + (q_rel >= E ? NR : 0)
__global__ void sid_kernel(const int* __restrict__ rel_ids,
                           const int* __restrict__ edge_type,
                           const int* __restrict__ q_rel,
                           int* __restrict__ esid, int* __restrict__ qsid) {
    long long t = (long long)blockIdx.x * 256 + threadIdx.x;
    if (t < Ee) esid[t] = rel_ids[edge_type[t]];
    if (t < Qq) {
        int qr   = q_rel[t];
        int base = (qr >= Ee) ? (qr - Ee) : qr;
        qsid[t]  = rel_ids[base] + ((qr >= Ee) ? NRr : 0);
    }
}

// qualifier scatter-sum: qsum[q_edge] += x_r_small[qsid[q]] * x_e[q_ent]
__global__ void qual_scatter_kernel(const float* __restrict__ xe,
                                    const float* __restrict__ xr,
                                    const int* __restrict__ qsid,
                                    const int* __restrict__ quali,
                                    float* __restrict__ qsum) {
    long long t = (long long)blockIdx.x * 256 + threadIdx.x;  // Qq*64
    int q  = (int)(t >> 6);
    int c  = ((int)t & 63) << 2;
    int qe = quali[2 * Qq + q];
    int qn = quali[Qq + q];
    int rid = qsid[q];
    float4 a = *(const float4*)(xr + (size_t)rid * Dd + c);
    float4 b = *(const float4*)(xe + (size_t)qn * Dd + c);
    float* d = qsum + (size_t)qe * Dd + c;
    atomicAdd(d + 0, a.x * b.x);
    atomicAdd(d + 1, a.y * b.y);
    atomicAdd(d + 2, a.z * b.z);
    atomicAdd(d + 3, a.w * b.w);
}

// Final sum-pooling: out[graph_ids[n]] += x_e[n]
__global__ void pool_kernel(const float* __restrict__ xe,
                            const int* __restrict__ gid,
                            float* __restrict__ out) {
    long long t = (long long)blockIdx.x * 256 + threadIdx.x;  // Nn*64
    int n = (int)(t >> 6);
    int c = ((int)t & 63) << 2;
    int g = gid[n];
    float4 v = *(const float4*)(xe + (size_t)n * Dd + c);
    float* d = out + (size_t)g * Dd + c;
    atomicAdd(d + 0, v.x);
    atomicAdd(d + 1, v.y);
    atomicAdd(d + 2, v.z);
    atomicAdd(d + 3, v.w);
}

// ---------------------------------------------------------------------------
// Fused per-edge-tile message kernel: 16 edges per workgroup, 8 waves.
//   qp   = qsum_tile @ W_q                          (WMMA, -> LDS)
//   cf   = x_e[src] * (a*x_r[rf] + (1-a)*qp)        (compose, -> LDS)
//   agg[dst] += cf @ W_in                           (WMMA + atomic scatter)
//   cb   = x_e[dst] * (a*x_r[rf+NR] + (1-a)*qp)
//   agg[src] += cb @ W_out
// ---------------------------------------------------------------------------
__global__ __launch_bounds__(256) void msg_kernel(
    const float* __restrict__ xe_cur, const float* __restrict__ xr_cur,
    const float* __restrict__ qsum, const int* __restrict__ edge_index,
    const int* __restrict__ esid,
    const v16bf* __restrict__ Wq, const v16bf* __restrict__ Win,
    const v16bf* __restrict__ Wout, float* __restrict__ agg) {
    __shared__ __align__(16) float Alds[16 * Dd];
    __shared__ __align__(16) float QP[16 * Dd];
    __shared__ int srcS[16], dstS[16], ridS[16];

    const int tid  = threadIdx.x;
    const int lane = tid & 31;
    const int wave = tid >> 5;
    const long long e0 = (long long)blockIdx.x * 16;

    // stage qsum tile into LDS (16 rows x 256 f32) via async global->LDS DMA
    {
        int i = tid >> 4, cb = (tid & 15) << 4;
        const float* gsrc = qsum + (e0 + i) * Dd + cb;
        float* ldst = Alds + i * Dd + cb;
#pragma unroll
        for (int u = 0; u < 4; ++u) {
            unsigned lds_addr = (unsigned)(unsigned long long)(ldst + u * 4);
            unsigned long long gaddr = (unsigned long long)(gsrc + u * 4);
            asm volatile("global_load_async_to_lds_b128 %0, %1, off"
                         :: "v"(lds_addr), "v"(gaddr) : "memory");
        }
    }
    if (tid < 16) {
        long long e = e0 + tid;
        srcS[tid] = edge_index[e];
        dstS[tid] = edge_index[Ee + e];
        ridS[tid] = esid[e];
    }
    asm volatile("s_wait_asynccnt 0x0" ::: "memory");
    __syncthreads();

    // GEMM1: qualifier projection -> QP
    {
        v8f acc0, acc1;
        tile_gemm_pair(Alds, Wq, wave * 2, wave * 2 + 1, lane, acc0, acc1);
        int colL = lane & 15;
        int mb   = (lane < 16) ? 0 : 8;
#pragma unroll
        for (int r = 0; r < 8; ++r) {
            QP[(mb + r) * Dd + (wave * 2) * 16 + colL]     = acc0[r];
            QP[(mb + r) * Dd + (wave * 2 + 1) * 16 + colL] = acc1[r];
        }
    }
    __syncthreads();

    // forward compose into Alds
    {
        int i = tid >> 4, cb = (tid & 15) << 4;
        const float4* xe4 = (const float4*)(xe_cur + (size_t)srcS[i] * Dd + cb);
        const float4* xr4 = (const float4*)(xr_cur + (size_t)ridS[i] * Dd + cb);
        const float4* qp4 = (const float4*)(QP + i * Dd + cb);
        float4* a4 = (float4*)(Alds + i * Dd + cb);
#pragma unroll
        for (int u = 0; u < 4; ++u) {
            float4 xe = xe4[u], xr = xr4[u], qp = qp4[u], o;
            o.x = xe.x * (ALPHA_F * xr.x + (1.f - ALPHA_F) * qp.x);
            o.y = xe.y * (ALPHA_F * xr.y + (1.f - ALPHA_F) * qp.y);
            o.z = xe.z * (ALPHA_F * xr.z + (1.f - ALPHA_F) * qp.z);
            o.w = xe.w * (ALPHA_F * xr.w + (1.f - ALPHA_F) * qp.w);
            a4[u] = o;
        }
    }
    __syncthreads();

    // GEMM2: forward messages -> atomic scatter onto agg[dst]
    {
        v8f acc0, acc1;
        tile_gemm_pair(Alds, Win, wave * 2, wave * 2 + 1, lane, acc0, acc1);
        int colL = lane & 15;
        int mb   = (lane < 16) ? 0 : 8;
#pragma unroll
        for (int r = 0; r < 8; ++r) {
            size_t rowoff = (size_t)dstS[mb + r] * Dd;
            atomicAdd(agg + rowoff + (wave * 2) * 16 + colL, acc0[r]);
            atomicAdd(agg + rowoff + (wave * 2 + 1) * 16 + colL, acc1[r]);
        }
    }
    __syncthreads();

    // backward compose into Alds
    {
        int i = tid >> 4, cb = (tid & 15) << 4;
        const float4* xe4 = (const float4*)(xe_cur + (size_t)dstS[i] * Dd + cb);
        const float4* xr4 = (const float4*)(xr_cur + (size_t)(ridS[i] + NRr) * Dd + cb);
        const float4* qp4 = (const float4*)(QP + i * Dd + cb);
        float4* a4 = (float4*)(Alds + i * Dd + cb);
#pragma unroll
        for (int u = 0; u < 4; ++u) {
            float4 xe = xe4[u], xr = xr4[u], qp = qp4[u], o;
            o.x = xe.x * (ALPHA_F * xr.x + (1.f - ALPHA_F) * qp.x);
            o.y = xe.y * (ALPHA_F * xr.y + (1.f - ALPHA_F) * qp.y);
            o.z = xe.z * (ALPHA_F * xr.z + (1.f - ALPHA_F) * qp.z);
            o.w = xe.w * (ALPHA_F * xr.w + (1.f - ALPHA_F) * qp.w);
            a4[u] = o;
        }
    }
    __syncthreads();

    // GEMM3: backward messages -> atomic scatter onto agg[src]
    {
        v8f acc0, acc1;
        tile_gemm_pair(Alds, Wout, wave * 2, wave * 2 + 1, lane, acc0, acc1);
        int colL = lane & 15;
        int mb   = (lane < 16) ? 0 : 8;
#pragma unroll
        for (int r = 0; r < 8; ++r) {
            size_t rowoff = (size_t)srcS[mb + r] * Dd;
            atomicAdd(agg + rowoff + (wave * 2) * 16 + colL, acc0[r]);
            atomicAdd(agg + rowoff + (wave * 2 + 1) * 16 + colL, acc1[r]);
        }
    }
}

// ---------------------------------------------------------------------------
// Self-loop message + finalize: x_e_new = relu((agg + (x_e*loop)@W_loop)/3)
// One 16-node tile per workgroup; exclusive tile ownership -> no atomics.
// ---------------------------------------------------------------------------
__global__ __launch_bounds__(256) void loop_finalize_kernel(
    const float* __restrict__ xe_cur, const float* __restrict__ loop_rel_l,
    const v16bf* __restrict__ Wloop, float* __restrict__ agg) {
    __shared__ __align__(16) float Alds[16 * Dd];
    const int tid = threadIdx.x, lane = tid & 31, wave = tid >> 5;
    const long long n0 = (long long)blockIdx.x * 16;
    {
        int i = tid >> 4, cb = (tid & 15) << 4;
        const float4* s = (const float4*)(xe_cur + (n0 + i) * Dd + cb);
        const float4* lr = (const float4*)(loop_rel_l + cb);
        float4* d = (float4*)(Alds + i * Dd + cb);
#pragma unroll
        for (int u = 0; u < 4; ++u) {
            float4 a = s[u], b = lr[u], o;
            o.x = a.x * b.x; o.y = a.y * b.y; o.z = a.z * b.z; o.w = a.w * b.w;
            d[u] = o;
        }
    }
    __syncthreads();
    {
        v8f acc0, acc1;
        tile_gemm_pair(Alds, Wloop, wave * 2, wave * 2 + 1, lane, acc0, acc1);
        int colL = lane & 15;
        int mb   = (lane < 16) ? 0 : 8;
#pragma unroll
        for (int r = 0; r < 8; ++r) {
            size_t i0 = (size_t)(n0 + mb + r) * Dd + (wave * 2) * 16 + colL;
            size_t i1 = (size_t)(n0 + mb + r) * Dd + (wave * 2 + 1) * 16 + colL;
            float v0 = (agg[i0] + acc0[r]) * (1.0f / 3.0f);
            float v1 = (agg[i1] + acc1[r]) * (1.0f / 3.0f);
            agg[i0] = v0 > 0.f ? v0 : 0.f;
            agg[i1] = v1 > 0.f ? v1 : 0.f;
        }
    }
}

// x_r_small_new = x_r_small @ W_rel  (1000 rows)
__global__ __launch_bounds__(256) void xr_update_kernel(
    const float* __restrict__ xr, const v16bf* __restrict__ Wrel,
    float* __restrict__ xro, int rows) {
    __shared__ __align__(16) float Alds[16 * Dd];
    const int tid = threadIdx.x, lane = tid & 31, wave = tid >> 5;
    const int r0 = blockIdx.x * 16;
    {
        int i = tid >> 4, cb = (tid & 15) << 4;
        int row = r0 + i;
        float4* d = (float4*)(Alds + i * Dd + cb);
        if (row < rows) {
            const float4* s = (const float4*)(xr + (size_t)row * Dd + cb);
            d[0] = s[0]; d[1] = s[1]; d[2] = s[2]; d[3] = s[3];
        } else {
            float4 z = make_float4(0.f, 0.f, 0.f, 0.f);
            d[0] = z; d[1] = z; d[2] = z; d[3] = z;
        }
    }
    __syncthreads();
    {
        v8f acc0, acc1;
        tile_gemm_pair(Alds, Wrel, wave * 2, wave * 2 + 1, lane, acc0, acc1);
        int colL = lane & 15;
        int mb   = (lane < 16) ? 0 : 8;
#pragma unroll
        for (int r = 0; r < 8; ++r) {
            int row = r0 + mb + r;
            if (row < rows) {
                xro[(size_t)row * Dd + (wave * 2) * 16 + colL]     = acc0[r];
                xro[(size_t)row * Dd + (wave * 2 + 1) * 16 + colL] = acc1[r];
            }
        }
    }
}

// ---------------------------------------------------------------------------
// Host launcher
// ---------------------------------------------------------------------------
extern "C" void kernel_launch(void* const* d_in, const int* in_sizes, int n_in,
                              void* d_out, int out_size, void* d_ws, size_t ws_size,
                              hipStream_t stream) {
    (void)in_sizes; (void)n_in; (void)out_size; (void)ws_size;
    const float* x_e_table    = (const float*)d_in[0];
    const float* x_r_table    = (const float*)d_in[1];
    const float* W_in_f       = (const float*)d_in[2];
    const float* W_out_f      = (const float*)d_in[3];
    const float* W_loop_f     = (const float*)d_in[4];
    const float* W_rel_f      = (const float*)d_in[5];
    const float* W_q_f        = (const float*)d_in[6];
    const float* loop_rel     = (const float*)d_in[7];
    const int*   entity_ids   = (const int*)d_in[8];
    const int*   relation_ids = (const int*)d_in[9];
    const int*   edge_index   = (const int*)d_in[10];
    const int*   edge_type    = (const int*)d_in[11];
    const int*   quali        = (const int*)d_in[12];
    const int*   graph_ids    = (const int*)d_in[13];
    float*       out          = (float*)d_out;

    // workspace layout
    char* ws = (char*)d_ws;
    const size_t SZ_XE   = (size_t)Nn * Dd * 4;   // 128 MB
    const size_t SZ_QS   = (size_t)Ee * Dd * 4;   // 256 MB
    float*  xeA   = (float*)(ws + 0);
    float*  xeB   = (float*)(ws + SZ_XE);
    float*  qsum  = (float*)(ws + 2 * SZ_XE);
    float*  xrW   = (float*)(ws + 2 * SZ_XE + SZ_QS);                 // 1000x256 f32
    __bf16* wprep = (__bf16*)(ws + 2 * SZ_XE + SZ_QS + (1 << 20));    // 10 * 65536 bf16
    int*    esid  = (int*)(ws + 2 * SZ_XE + SZ_QS + (1 << 20) + 10 * 65536 * 2);
    int*    qsid  = esid + Ee;

    const int WSZ = Dd * Dd;  // 65536 elements per weight matrix
    // prep weights: layer-major slots [Wq, Win, Wout, Wloop, Wrel]
    for (int l = 0; l < LL; ++l) {
        prep_w_kernel<<<WSZ / 256, 256, 0, stream>>>(W_q_f    + (size_t)l * WSZ, wprep + (size_t)(l * 5 + 0) * WSZ);
        prep_w_kernel<<<WSZ / 256, 256, 0, stream>>>(W_in_f   + (size_t)l * WSZ, wprep + (size_t)(l * 5 + 1) * WSZ);
        prep_w_kernel<<<WSZ / 256, 256, 0, stream>>>(W_out_f  + (size_t)l * WSZ, wprep + (size_t)(l * 5 + 2) * WSZ);
        prep_w_kernel<<<WSZ / 256, 256, 0, stream>>>(W_loop_f + (size_t)l * WSZ, wprep + (size_t)(l * 5 + 3) * WSZ);
        prep_w_kernel<<<WSZ / 256, 256, 0, stream>>>(W_rel_f  + (size_t)l * WSZ, wprep + (size_t)(l * 5 + 4) * WSZ);
    }

    gather_xe_kernel<<<(Nn * 64) / 256, 256, 0, stream>>>(x_e_table, entity_ids, xeA);
    sid_kernel<<<Qq / 256, 256, 0, stream>>>(relation_ids, edge_type, quali, esid, qsid);

    const float* xr_cur = x_r_table;  // 1000 x 256 distinct relation rows
    for (int l = 0; l < LL; ++l) {
        float* cur = (l == 0) ? xeA : xeB;
        float* agg = (l == 0) ? xeB : xeA;
        const v16bf* Wq    = (const v16bf*)(wprep + (size_t)(l * 5 + 0) * WSZ);
        const v16bf* Win   = (const v16bf*)(wprep + (size_t)(l * 5 + 1) * WSZ);
        const v16bf* Wout  = (const v16bf*)(wprep + (size_t)(l * 5 + 2) * WSZ);
        const v16bf* Wloop = (const v16bf*)(wprep + (size_t)(l * 5 + 3) * WSZ);
        const v16bf* Wrel  = (const v16bf*)(wprep + (size_t)(l * 5 + 4) * WSZ);

        zero_kernel<<<(int)(((size_t)Ee * Dd / 4 + 255) / 256), 256, 0, stream>>>((float4*)qsum, (long long)Ee * Dd / 4);
        zero_kernel<<<(int)(((size_t)Nn * Dd / 4 + 255) / 256), 256, 0, stream>>>((float4*)agg, (long long)Nn * Dd / 4);

        qual_scatter_kernel<<<(Qq * 64) / 256, 256, 0, stream>>>(cur, xr_cur, qsid, quali, qsum);
        msg_kernel<<<Ee / 16, 256, 0, stream>>>(cur, xr_cur, qsum, edge_index, esid, Wq, Win, Wout, agg);
        loop_finalize_kernel<<<Nn / 16, 256, 0, stream>>>(cur, loop_rel + (size_t)l * Dd, Wloop, agg);

        if (l + 1 < LL) {
            xr_update_kernel<<<(2 * NRr + 15) / 16, 256, 0, stream>>>(xr_cur, Wrel, xrW, 2 * NRr);
            xr_cur = xrW;
        }
    }

    zero_kernel<<<(int)(((size_t)Gg * Dd / 4 + 255) / 256), 256, 0, stream>>>((float4*)out, (long long)Gg * Dd / 4);
    // after layer 1 the finalized x_e lives in xeA (agg of last layer)
    pool_kernel<<<(Nn * 64) / 256, 256, 0, stream>>>(xeA, graph_ids, out);
}
